// BasicBlock_6983616824333
// MI455X (gfx1250) — compile-verified, matrix operand order checked by reading, not run on hardware
//
#include <hip/hip_runtime.h>
#include <cstdint>

typedef __attribute__((ext_vector_type(16))) int   v16i;
typedef __attribute__((ext_vector_type(8)))  float v8f;

#define BB    64
#define CC    256
#define HH    28
#define WW_   28
#define HWP   784        // 28*28
#define PP    50176      // 64*784 output positions
#define HP    30         // padded height
#define WP    30         // padded width
#define HWPAD 900        // 30*30 padded positions per image
// GEMM: M = PP, N = CC, K = CC*9. 16x16 tiles; K = 9 taps x 2 blocks of 128 (FP8 WMMA).
// K (ci) dimension is stored PERMUTED so A operands are contiguous b128 chunks:
// within each 64-K block, 8B chunks ordered {0,2,1,3,4,6,5,7} (swap 1<->2, 5<->6).

// FP8 E4M3 encodings of sign(): +1 -> 0x38, -1 -> 0xB8, 0 -> 0x00 (exact values)
__device__ __forceinline__ uint32_t fp8sign(float v) {
    return v > 0.f ? 0x38u : (v < 0.f ? 0xB8u : 0u);
}

// K-permutation applied to channel index (both activations and weights)
__device__ __forceinline__ int kperm(int ci) {
    int sub = (ci >> 3) & 7;
    int s2  = sub & 3;
    int slot = (s2 == 1 || s2 == 2) ? (sub ^ 3) : sub;
    return (ci & ~63) | (slot << 3) | (ci & 7);
}

// ---------------- init: zero the stats region (8*256 floats) ----------------
__global__ void k_init_stats(float* stats) {
    int i = blockIdx.x * blockDim.x + threadIdx.x;
    if (i < 2048) stats[i] = 0.f;
}

// ------- quantize x: NCHW f32 -> zero-padded NHWC fp8 [64][30][30][256], K-permuted -------
__global__ void k_quant_x(const float* __restrict__ x, int8_t* __restrict__ xq) {
    int idx4 = blockIdx.x * blockDim.x + threadIdx.x;   // one thread = 4 channels
    if (idx4 >= BB * HWPAD * (CC / 4)) return;
    int pp  = idx4 >> 6;            // padded position (n*900 + hp*30 + wp)
    int c4  = (idx4 & 63) << 2;     // channel base (source order)
    int n   = pp / HWPAD;
    int rem = pp % HWPAD;
    int hp  = rem / WP;
    int wp  = rem % WP;
    uint32_t packed = 0;
    if (hp >= 1 && hp <= HH && wp >= 1 && wp <= WW_) {
        int hw = (hp - 1) * WW_ + (wp - 1);
        const float* src = x + (size_t)n * CC * HWP + hw;
#pragma unroll
        for (int i = 0; i < 4; ++i) {
            float v = src[(size_t)(c4 + i) * HWP];
            packed |= fp8sign(v) << (8 * i);
        }
    }
    // 4-aligned groups stay inside one 8-chunk, so the permutation moves the whole dword
    *(uint32_t*)(xq + (size_t)pp * CC + kperm(c4)) = packed;
}

// ---------------- quantize w: OIHW f32 -> [tap][co][perm(ci)] fp8 ----------------
__global__ void k_quant_w(const float* __restrict__ w, int8_t* __restrict__ wq) {
    int idx = blockIdx.x * blockDim.x + threadIdx.x;    // idx = tap*65536 + co*256 + ci
    if (idx >= 9 * CC * CC) return;
    int tap = idx >> 16;
    int rem = idx & 65535;
    int co  = rem >> 8;
    int ci  = rem & 255;
    float v = w[(size_t)co * CC * 9 + (size_t)ci * 9 + tap];
    wq[(tap << 16) + (co << 8) + kperm(ci)] = (int8_t)fp8sign(v);
}

// -------- binary 3x3 conv via implicit GEMM, FP8 16x16x128 WMMA --------
// Each wave computes TWO 16x16 output tiles sharing one B operand (2 accumulators,
// independent WMMAs fill each other's hazard slots). A loads are 4x b128 per WMMA
// thanks to the K-permuted layout; pad rows are fp8 zero so no border logic.
__global__ __launch_bounds__(256) void k_bconv(const int8_t* __restrict__ Aq,
                                               const int8_t* __restrict__ Wq,
                                               const float*  __restrict__ bias,
                                               float*        __restrict__ Y,
                                               float*        __restrict__ sum,
                                               float*        __restrict__ sumsq) {
    __shared__ int4 ldsW[2304];                 // 9 taps * 16 co * 256 ci bytes = 36 KB
    const int tid    = threadIdx.x;
    const int coBase = blockIdx.y * 16;

    // stage B tile: global [tap][co][ci'] -> LDS [tap][col][ci'], 16B chunks
    const int4* wsrc = (const int4*)Wq;
#pragma unroll
    for (int i = 0; i < 9; ++i) {
        int q   = tid + i * 256;                // 0..2303
        int tap = q >> 8;
        int rem = q & 255;
        int col = rem >> 4;
        int k16 = rem & 15;
        ldsW[q] = wsrc[(size_t)(tap * CC + coBase + col) * 16 + k16];
    }
    __syncthreads();

    const int  lane   = tid & 31;
    const int  wave   = tid >> 5;
    const bool laneHi = lane >= 16;
    const int  r      = lane & 15;              // A row / B column / D column index
    const int  hi16   = laneHi ? 16 : 0;        // A-layout byte offset for upper half-wave
    const int  hi1    = laneHi ? 1 : 0;         // B-layout chunk offset
    const int  mBase  = (blockIdx.x * 8 + wave) * 32;   // two 16-row tiles per wave

    // per-tile padded base pointers: element (n, h+dy, w+dx) = base + (dy*30+dx)*256
    const int p0 = mBase + r;
    const int p1 = mBase + 16 + r;
    const int n0 = p0 / HWP, hw0 = p0 % HWP;
    const int n1 = p1 / HWP, hw1 = p1 % HWP;
    const int8_t* abase0 =
        Aq + (((size_t)n0 * HWPAD + (size_t)(hw0 / WW_) * WP + (hw0 % WW_)) << 8);
    const int8_t* abase1 =
        Aq + (((size_t)n1 * HWPAD + (size_t)(hw1 / WW_) * WP + (hw1 % WW_)) << 8);
    const int ldsCol = r << 4;

    v8f acc0 = {0.f, 0.f, 0.f, 0.f, 0.f, 0.f, 0.f, 0.f};
    v8f acc1 = {0.f, 0.f, 0.f, 0.f, 0.f, 0.f, 0.f, 0.f};

#pragma unroll
    for (int dy = 0; dy < 3; ++dy) {
#pragma unroll
        for (int dx = 0; dx < 3; ++dx) {
            const int tap    = dy * 3 + dx;
            const int tapOff = (dy * WP + dx) << 8;    // compile-time per tap
#pragma unroll
            for (int kb = 0; kb < 2; ++kb) {           // two K=128 blocks cover ci'=0..255
                // B 128x16 fp8: lane = column; 16B chunks at g*32 + (laneHi?16:0)
                v16i b;
                const int li = tap * 256 + ldsCol + kb * 8 + hi1;
#pragma unroll
                for (int g = 0; g < 4; ++g) {
                    int4 q = ldsW[li + g * 2];
                    b[g * 4 + 0] = q.x; b[g * 4 + 1] = q.y;
                    b[g * 4 + 2] = q.z; b[g * 4 + 3] = q.w;
                }
                // A (K-permuted): per lane 4 contiguous b128 at kb*128 + hi16 + {0,32,64,96}
                const int aoff = tapOff + kb * 128 + hi16;
                v16i a0, a1;
#pragma unroll
                for (int g = 0; g < 4; ++g) {
                    int4 q = *(const int4*)(abase0 + aoff + g * 32);
                    a0[g * 4 + 0] = q.x; a0[g * 4 + 1] = q.y;
                    a0[g * 4 + 2] = q.z; a0[g * 4 + 3] = q.w;
                }
#pragma unroll
                for (int g = 0; g < 4; ++g) {
                    int4 q = *(const int4*)(abase1 + aoff + g * 32);
                    a1[g * 4 + 0] = q.x; a1[g * 4 + 1] = q.y;
                    a1[g * 4 + 2] = q.z; a1[g * 4 + 3] = q.w;
                }
                acc0 = __builtin_amdgcn_wmma_f32_16x16x128_fp8_fp8(a0, b, (short)0, acc0,
                                                                   false, false);
                acc1 = __builtin_amdgcn_wmma_f32_16x16x128_fp8_fp8(a1, b, (short)0, acc1,
                                                                   false, false);
            }
        }
    }

    // epilogue: bias add, f32 store (NHWC), per-channel BN partial sums (both tiles)
    const float bv     = bias[coBase + r];
    const int   rowAdd = laneHi ? 8 : 0;
    float s = 0.f, s2 = 0.f;
#pragma unroll
    for (int rr = 0; rr < 8; ++rr) {
        float v0 = acc0[rr] + bv;
        float v1 = acc1[rr] + bv;
        Y[(size_t)(mBase + rr + rowAdd) * CC + coBase + r]      = v0;
        Y[(size_t)(mBase + 16 + rr + rowAdd) * CC + coBase + r] = v1;
        s  += v0 + v1;
        s2 += v0 * v0 + v1 * v1;
    }
    s  += __shfl_xor(s, 16, 32);
    s2 += __shfl_xor(s2, 16, 32);
    if (lane < 16) {
        atomicAdd(&sum[coBase + r], s);
        atomicAdd(&sumsq[coBase + r], s2);
    }
}

// ---------------- per-channel BN prep: mean, gamma*rsqrt(var+eps) ----------------
__global__ void k_bn_prep(const float* __restrict__ sum, const float* __restrict__ sumsq,
                          const float* __restrict__ gamma,
                          float* __restrict__ mean_out, float* __restrict__ inv_out) {
    int c = threadIdx.x;
    float m   = sum[c] * (1.f / (float)PP);
    float var = sumsq[c] * (1.f / (float)PP) - m * m;
    mean_out[c] = m;
    inv_out[c]  = gamma[c] * rsqrtf(var + 1e-5f);
}

// -------- BN + sign -> padded fp8 NHWC (K-permuted; feeds conv2) --------
__global__ void k_binarize(const float* __restrict__ Y, const float* __restrict__ mean,
                           const float* __restrict__ inv, const float* __restrict__ beta,
                           int8_t* __restrict__ xq) {
    int idx4 = blockIdx.x * blockDim.x + threadIdx.x;
    if (idx4 >= BB * HWPAD * (CC / 4)) return;
    int pp  = idx4 >> 6;
    int c4  = (idx4 & 63) << 2;
    int n   = pp / HWPAD;
    int rem = pp % HWPAD;
    int hp  = rem / WP;
    int wp  = rem % WP;
    uint32_t packed = 0;
    if (hp >= 1 && hp <= HH && wp >= 1 && wp <= WW_) {
        int p = n * HWP + (hp - 1) * WW_ + (wp - 1);
#pragma unroll
        for (int i = 0; i < 4; ++i) {
            int c = c4 + i;
            float t = (Y[(size_t)p * CC + c] - mean[c]) * inv[c] + beta[c];
            packed |= fp8sign(t) << (8 * i);
        }
    }
    *(uint32_t*)(xq + (size_t)pp * CC + kperm(c4)) = packed;
}

// ---------------- BN2 + residual, NHWC f32 -> NCHW f32 output ----------------
__global__ void k_finalize(const float* __restrict__ Y, const float* __restrict__ mean,
                           const float* __restrict__ inv, const float* __restrict__ beta,
                           const float* __restrict__ x, float* __restrict__ out) {
    int idx = blockIdx.x * blockDim.x + threadIdx.x;
    if (idx >= BB * CC * HWP) return;
    int n   = idx / (CC * HWP);
    int rem = idx % (CC * HWP);
    int c   = rem / HWP;
    int hw  = rem % HWP;
    int p   = n * HWP + hw;
    out[idx] = (Y[(size_t)p * CC + c] - mean[c]) * inv[c] + beta[c] + x[idx];
}

extern "C" void kernel_launch(void* const* d_in, const int* in_sizes, int n_in,
                              void* d_out, int out_size, void* d_ws, size_t ws_size,
                              hipStream_t stream) {
    const float* x   = (const float*)d_in[0];
    const float* w1  = (const float*)d_in[1];
    const float* b1  = (const float*)d_in[2];
    const float* g1  = (const float*)d_in[3];
    const float* be1 = (const float*)d_in[4];
    const float* w2  = (const float*)d_in[5];
    const float* b2  = (const float*)d_in[6];
    const float* g2  = (const float*)d_in[7];
    const float* be2 = (const float*)d_in[8];
    float* out = (float*)d_out;

    // workspace layout (256B aligned): Y f32 NHWC | xq pad | x2q pad | wq1 | wq2 | stats
    char*   ws   = (char*)d_ws;
    float*  Y    = (float*) (ws);                         // 51,380,224 B
    int8_t* xq   = (int8_t*)(ws + 51380224);              // 14,745,600 B (padded)
    int8_t* x2q  = (int8_t*)(ws + 66125824);              // 14,745,600 B (padded)
    int8_t* wq1  = (int8_t*)(ws + 80871424);              //    589,824 B
    int8_t* wq2  = (int8_t*)(ws + 81461248);              //    589,824 B
    float*  st   = (float*) (ws + 82051072);              //      8,192 B
    float *sum1 = st, *sq1 = st + 256, *sum2 = st + 512, *sq2 = st + 768;
    float *mean1 = st + 1024, *inv1 = st + 1280, *mean2 = st + 1536, *inv2 = st + 1792;

    const int padBlocks = (BB * HWPAD * (CC / 4)) / 256;  // 14400
    dim3 convGrid(196, 16);           // 196 blocks * 8 waves * 2 tiles * 16 rows = 50176 M

    k_init_stats<<<8, 256, 0, stream>>>(st);
    k_quant_x  <<<padBlocks, 256, 0, stream>>>(x, xq);
    k_quant_w  <<<(9 * CC * CC) / 256, 256, 0, stream>>>(w1, wq1);
    k_quant_w  <<<(9 * CC * CC) / 256, 256, 0, stream>>>(w2, wq2);

    k_bconv    <<<convGrid, 256, 0, stream>>>(xq, wq1, b1, Y, sum1, sq1);
    k_bn_prep  <<<1, 256, 0, stream>>>(sum1, sq1, g1, mean1, inv1);
    k_binarize <<<padBlocks, 256, 0, stream>>>(Y, mean1, inv1, be1, x2q);

    k_bconv    <<<convGrid, 256, 0, stream>>>(x2q, wq2, b2, Y, sum2, sq2);
    k_bn_prep  <<<1, 256, 0, stream>>>(sum2, sq2, g2, mean2, inv2);
    k_finalize <<<(BB * CC * HWP + 255) / 256, 256, 0, stream>>>(Y, mean2, inv2, be2, x, out);
}
